// GCN_64836826300546
// MI455X (gfx1250) — matrix operand order repaired and checked by reading, not compiled, not run on hardware
//
#include <hip/hip_runtime.h>
#include <hip/hip_bf16.h>

typedef __attribute__((ext_vector_type(16))) __bf16 v16bf;
typedef __attribute__((ext_vector_type(8)))  __bf16 v8bf;
typedef __attribute__((ext_vector_type(8)))  float  v8f;
typedef int v4i __attribute__((vector_size(16)));   // matches async-LDS builtin param

#if __has_builtin(__builtin_amdgcn_global_load_async_to_lds_b128) && \
    __has_builtin(__builtin_amdgcn_s_wait_asynccnt)
#define HAVE_ASYNC_LDS 1
#else
#define HAVE_ASYNC_LDS 0
#endif

__device__ __forceinline__ v16bf cat16(v8bf a, v8bf b) {
    return __builtin_shufflevector(a, b, 0, 1, 2, 3, 4, 5, 6, 7,
                                   8, 9, 10, 11, 12, 13, 14, 15);
}

// ---------------------------------------------------------------------------
// Small elementwise helpers
// ---------------------------------------------------------------------------
__global__ void fill_kernel(float* __restrict__ p, float v, size_t n) {
    size_t i = (size_t)blockIdx.x * blockDim.x + threadIdx.x;
    if (i < n) p[i] = v;
}

__global__ void deg_accum_kernel(const int* __restrict__ col,
                                 const float* __restrict__ ew,
                                 float* __restrict__ deg, int E) {
    int e = blockIdx.x * blockDim.x + threadIdx.x;
    if (e < E) atomicAdd(&deg[col[e]], ew[e]);
}

__global__ void rsqrt_kernel(float* __restrict__ deg, int N) {
    int i = blockIdx.x * blockDim.x + threadIdx.x;
    if (i < N) {
        float d = deg[i];
        deg[i] = (d > 0.0f) ? rsqrtf(d) : 0.0f;   // becomes dinv in place
    }
}

__global__ void edge_norm_kernel(const int* __restrict__ row,
                                 const int* __restrict__ col,
                                 const float* __restrict__ ew,
                                 const float* __restrict__ dinv,
                                 float* __restrict__ nrm, int E) {
    int e = blockIdx.x * blockDim.x + threadIdx.x;
    if (e < E) nrm[e] = dinv[row[e]] * ew[e] * dinv[col[e]];
}

// fp32 -> bf16 hi/lo split (rowwise, layout-preserving)
__global__ void split_rows_kernel(const float* __restrict__ in,
                                  __bf16* __restrict__ hi,
                                  __bf16* __restrict__ lo, size_t n) {
    size_t i = (size_t)blockIdx.x * blockDim.x + threadIdx.x;
    if (i < n) {
        float v = in[i];
        __bf16 h = (__bf16)v;
        hi[i] = h;
        lo[i] = (__bf16)(v - (float)h);
    }
}

// W[R][Cc] fp32 -> Wt[Cc][R] bf16 hi/lo (transpose + split; tiny matrix)
__global__ void split_transpose_kernel(const float* __restrict__ W,
                                       __bf16* __restrict__ wthi,
                                       __bf16* __restrict__ wtlo,
                                       int R, int Cc) {
    int i = blockIdx.x * blockDim.x + threadIdx.x;
    if (i < R * Cc) {
        int r = i / Cc, c = i - r * Cc;
        float v = W[i];
        __bf16 h = (__bf16)v;
        wthi[(size_t)c * R + r] = h;
        wtlo[(size_t)c * R + r] = (__bf16)(v - (float)h);
    }
}

// ---------------------------------------------------------------------------
// Dense GEMM  C[N,F_OUT] = A[N,128] @ W[128,F_OUT]  via bf16-split WMMA.
// Inputs are pre-split bf16: Ahi/Alo row-major [N,128]; Wthi/Wtlo transposed
// [F_OUT][K].  W tiles are staged into padded LDS via the CDNA5 async
// global->LDS path (ASYNCcnt), shared by all 8 waves per block.  One wave32
// -> one 16x16 output tile, K=128 in four K=32 steps, 3 WMMAs/step
// (hi*hi + lo*hi + hi*lo ~ fp32 accuracy).
// VGPR layouts used here (contiguous k, same for A and B):
//   A 16x32 bf16 : lanes 0-15 row M=lane, K = kk+0..15;
//                  lanes 16-31 row M=lane-16, K = kk+16..31.
//   B 32x16 bf16 : lane col N=lane&15; lanes 0-15 K=kk+0..15, 16-31 K=kk+16..31.
//   C/D f32      : elem j -> M = j + 8*(lane>=16), N = lane&15.
// ---------------------------------------------------------------------------
template <int F_OUT>
__global__ void gemm_wmma_bf16split(const __bf16* __restrict__ Ahi,
                                    const __bf16* __restrict__ Alo,
                                    const __bf16* __restrict__ Wthi,
                                    const __bf16* __restrict__ Wtlo,
                                    float* __restrict__ C, int N) {
    constexpr int K = 128;
    constexpr int PITCH = K + 8;   // +16B/row: keeps b128 alignment, rotates banks
    constexpr int tilesN = F_OUT / 16;

    extern __shared__ __align__(16) __bf16 smem[];
    __bf16* sWhi = smem;
    __bf16* sWlo = smem + F_OUT * PITCH;

    // ---- stage Wt hi/lo into padded LDS ----
    constexpr int CH = (F_OUT * K) / 8;   // 16-byte chunks
    constexpr int CPR = K / 8;            // chunks per row
    for (int c = threadIdx.x; c < CH; c += blockDim.x) {
        const int r = c / CPR;
        const int co = (c - r * CPR) * 8;
        const size_t g = (size_t)r * K + co;
        const size_t l = (size_t)r * PITCH + co;
#if HAVE_ASYNC_LDS
        __builtin_amdgcn_global_load_async_to_lds_b128(
            (v4i*)const_cast<__bf16*>(Wthi + g), (v4i*)(sWhi + l), 0, 0);
        __builtin_amdgcn_global_load_async_to_lds_b128(
            (v4i*)const_cast<__bf16*>(Wtlo + g), (v4i*)(sWlo + l), 0, 0);
#else
        *(v8bf*)(sWhi + l) = *(const v8bf*)(Wthi + g);
        *(v8bf*)(sWlo + l) = *(const v8bf*)(Wtlo + g);
#endif
    }
#if HAVE_ASYNC_LDS
    __builtin_amdgcn_s_wait_asynccnt(0);
#endif
    __syncthreads();

    const int waveId = (int)((blockIdx.x * blockDim.x + threadIdx.x) >> 5);
    const int lane = threadIdx.x & 31;
    const int tilesM = (N + 15) >> 4;
    const int tm = waveId / tilesN;
    const int tn = waveId % tilesN;
    const int tmc = (tm < tilesM) ? tm : (tilesM - 1);  // clamp, keep EXEC all-1
    const int half = lane >> 4;
    const int l15 = lane & 15;
    int row = tmc * 16 + l15;
    if (row >= N) row = N - 1;
    const int ncol = tn * 16 + l15;

    const __bf16* __restrict__ aH = Ahi + (size_t)row * K;
    const __bf16* __restrict__ aL = Alo + (size_t)row * K;
    const __bf16* bH = sWhi + (size_t)ncol * PITCH + half * 16;
    const __bf16* bL = sWlo + (size_t)ncol * PITCH + half * 16;

    v8f acc = {};
#pragma unroll
    for (int kk = 0; kk < K; kk += 32) {
        const int k0 = kk + half * 16;
        const int k1 = kk + half * 16 + 8;
        v16bf ah = cat16(*(const v8bf*)(aH + k0), *(const v8bf*)(aH + k1));
        v16bf al = cat16(*(const v8bf*)(aL + k0), *(const v8bf*)(aL + k1));
        v16bf bh = cat16(*(const v8bf*)(bH + kk), *(const v8bf*)(bH + kk + 8));
        v16bf bl = cat16(*(const v8bf*)(bL + kk), *(const v8bf*)(bL + kk + 8));
        acc = __builtin_amdgcn_wmma_f32_16x16x32_bf16(false, ah, false, bh,
                                                      (short)0, acc, false, false);
        acc = __builtin_amdgcn_wmma_f32_16x16x32_bf16(false, al, false, bh,
                                                      (short)0, acc, false, false);
        acc = __builtin_amdgcn_wmma_f32_16x16x32_bf16(false, ah, false, bl,
                                                      (short)0, acc, false, false);
    }

    const bool live = (tm < tilesM);
#pragma unroll
    for (int j = 0; j < 8; ++j) {
        const int m = tmc * 16 + j + 8 * half;
        if (live && m < N) C[(size_t)m * F_OUT + ncol] = acc[j];
    }
}

// ---------------------------------------------------------------------------
// Edge scatter: agg[col[e]][f] += norm[e] * h[row[e]][f]
// One lane per (edge, feature): coalesced gather, f32 atomic scatter.
// h (25.6MB) and agg (25.6MB) are L2-resident on MI455X (192MB L2).
// ---------------------------------------------------------------------------
template <int F, int LOGF>
__global__ void scatter_edges_kernel(const int* __restrict__ row,
                                     const int* __restrict__ col,
                                     const float* __restrict__ nrm,
                                     const float* __restrict__ h,
                                     float* __restrict__ agg, int E) {
    size_t t = (size_t)blockIdx.x * blockDim.x + threadIdx.x;
    size_t e = t >> LOGF;
    int f = (int)(t & (F - 1));
    if (e >= (size_t)E) return;
    int r = row[e], c = col[e];
    float v = nrm[e] * h[(size_t)r * F + f];
    atomicAdd(&agg[(size_t)c * F + f], v);
}

// ---------------------------------------------------------------------------
// Finalize: out = [relu](agg + dinv^2 * h + bias); optional fused bf16 split
// of the result (feeds next layer's WMMA GEMM without another pass).
// ---------------------------------------------------------------------------
template <int F, int LOGF, bool RELU, bool SPLIT>
__global__ void finalize_kernel(const float* __restrict__ agg,
                                const float* __restrict__ h,
                                const float* __restrict__ dinv,
                                const float* __restrict__ bias,
                                float* __restrict__ out,
                                __bf16* __restrict__ hi,
                                __bf16* __restrict__ lo, int N) {
    size_t t = (size_t)blockIdx.x * blockDim.x + threadIdx.x;
    size_t n = t >> LOGF;
    int f = (int)(t & (F - 1));
    if (n >= (size_t)N) return;
    float dn = dinv[n];
    float v = agg[t] + dn * dn * h[t] + bias[f];
    if (RELU) v = fmaxf(v, 0.0f);
    out[t] = v;
    if (SPLIT) {
        __bf16 hv = (__bf16)v;
        hi[t] = hv;
        lo[t] = (__bf16)(v - (float)hv);
    }
}

// ---------------------------------------------------------------------------
// Orchestration
// ---------------------------------------------------------------------------
static inline unsigned int nblk(size_t n, int b) { return (unsigned int)((n + b - 1) / b); }

extern "C" void kernel_launch(void* const* d_in, const int* in_sizes, int n_in,
                              void* d_out, int out_size, void* d_ws, size_t ws_size,
                              hipStream_t stream) {
    (void)n_in; (void)out_size; (void)ws_size;
    constexpr int F_IN = 128, H = 128, Cdim = 64, K = 128;
    const float* x  = (const float*)d_in[0];
    const float* ew = (const float*)d_in[1];
    const float* W1 = (const float*)d_in[2];
    const float* b1 = (const float*)d_in[3];
    const float* W2 = (const float*)d_in[4];
    const float* b2 = (const float*)d_in[5];
    const int*  eidx = (const int*)d_in[6];

    const int N = in_sizes[0] / F_IN;
    const int E = in_sizes[1];
    const int* row = eidx;        // edge_index[0]
    const int* col = eidx + E;    // edge_index[1]

    // Workspace carve-out (256B-aligned blocks)
    char* wsp = (char*)d_ws;
    auto alloc = [&](size_t bytes) -> char* {
        char* p = wsp;
        wsp += (bytes + 255) & ~(size_t)255;
        return p;
    };
    float*  dinv = (float*)alloc((size_t)N * 4);
    float*  nrm  = (float*)alloc((size_t)E * 4);
    float*  hbuf = (float*)alloc((size_t)N * H * 4);
    float*  agg  = (float*)alloc((size_t)N * H * 4);
    __bf16* ahi  = (__bf16*)alloc((size_t)N * K * 2);
    __bf16* alo  = (__bf16*)alloc((size_t)N * K * 2);
    __bf16* wthi = (__bf16*)alloc((size_t)K * H * 2);
    __bf16* wtlo = (__bf16*)alloc((size_t)K * H * 2);

    // Output layout: embedding [N,128] then out [N,64]
    float* emb  = (float*)d_out;
    float* out2 = emb + (size_t)N * H;

    const int B = 256;

    // ---- normalization: deg = 1 (self loop) + sum ew over col; dinv = rsqrt ----
    fill_kernel<<<nblk(N, B), B, 0, stream>>>(dinv, 1.0f, (size_t)N);
    deg_accum_kernel<<<nblk(E, B), B, 0, stream>>>(col, ew, dinv, E);
    rsqrt_kernel<<<nblk(N, B), B, 0, stream>>>(dinv, N);
    edge_norm_kernel<<<nblk(E, B), B, 0, stream>>>(row, col, ew, dinv, nrm, E);

    // ---- one-time precision splits ----
    split_rows_kernel<<<nblk((size_t)N * K, B), B, 0, stream>>>(x, ahi, alo, (size_t)N * K);
    split_transpose_kernel<<<nblk(K * H, B), B, 0, stream>>>(W1, wthi, wtlo, K, H);

    // ---- layer 1: h = x @ W1 ; scatter ; emb = relu(...) (+fused split) ----
    {
        const size_t waves = (size_t)((N + 15) / 16) * (H / 16);
        const size_t smem = 2 * (size_t)H * (K + 8) * sizeof(__bf16);
        gemm_wmma_bf16split<H><<<nblk(waves * 32, B), B, smem, stream>>>(
            ahi, alo, wthi, wtlo, hbuf, N);
        fill_kernel<<<nblk((size_t)N * H, B), B, 0, stream>>>(agg, 0.0f, (size_t)N * H);
        scatter_edges_kernel<H, 7><<<nblk((size_t)E * H, B), B, 0, stream>>>(
            row, col, nrm, hbuf, agg, E);
        finalize_kernel<H, 7, true, true><<<nblk((size_t)N * H, B), B, 0, stream>>>(
            agg, hbuf, dinv, b1, emb, ahi, alo, N);
    }

    // ---- layer 2: h = emb @ W2 ; scatter ; out = agg + self + b2 ----
    {
        split_transpose_kernel<<<nblk(K * Cdim, B), B, 0, stream>>>(W2, wthi, wtlo, K, Cdim);
        const size_t waves = (size_t)((N + 15) / 16) * (Cdim / 16);
        const size_t smem = 2 * (size_t)Cdim * (K + 8) * sizeof(__bf16);
        gemm_wmma_bf16split<Cdim><<<nblk(waves * 32, B), B, smem, stream>>>(
            ahi, alo, wthi, wtlo, hbuf, N);
        fill_kernel<<<nblk((size_t)N * Cdim, B), B, 0, stream>>>(agg, 0.0f, (size_t)N * Cdim);
        scatter_edges_kernel<Cdim, 6><<<nblk((size_t)E * Cdim, B), B, 0, stream>>>(
            row, col, nrm, hbuf, agg, E);
        finalize_kernel<Cdim, 6, false, false><<<nblk((size_t)N * Cdim, B), B, 0, stream>>>(
            agg, hbuf, dinv, b2, out2, nullptr, nullptr, N);
    }
}
